// TestRNN_738734375261
// MI455X (gfx1250) — matrix operand order repaired and checked by reading, not compiled, not run on hardware
//
#include <hip/hip_runtime.h>

// ---------------------------------------------------------------------------
// Problem constants (from reference): B=64, T=2048, D=256, V=256
// ---------------------------------------------------------------------------
#define RB 64
#define RT 2048
#define RD 256
#define RV 256

// Flip to 0 if the gfx1250 assembler rejects global_load_async_to_lds_b128.
#define RNN_USE_ASYNC_STAGING 1

typedef __attribute__((ext_vector_type(16))) __bf16 v16bf;
typedef __attribute__((ext_vector_type(8)))  float  v8f;

// ---- bf16 <-> f32 helpers (bits stored as unsigned short) ------------------
__device__ __forceinline__ unsigned short f2bf(float f) {
  unsigned int u = __float_as_uint(f);
  return (unsigned short)((u + 0x7FFFu + ((u >> 16) & 1u)) >> 16);
}
__device__ __forceinline__ float bf2f(unsigned short s) {
  return __uint_as_float(((unsigned int)s) << 16);
}

// ---- fast transcendentals (gfx1250 hardware paths, guarded) ----------------
__device__ __forceinline__ float fast_tanh(float x) {
#if __has_builtin(__builtin_amdgcn_tanhf)
  return __builtin_amdgcn_tanhf(x);        // v_tanh_f32
#else
  return tanhf(x);
#endif
}
__device__ __forceinline__ float fast_sigmoid(float x) {
#if __has_builtin(__builtin_amdgcn_exp2f)
  float e = __builtin_amdgcn_exp2f(x * -1.442695041f);  // v_exp_f32
#else
  float e = __expf(-x);
#endif
#if __has_builtin(__builtin_amdgcn_rcpf)
  return __builtin_amdgcn_rcpf(1.0f + e);  // v_rcp_f32
#else
  return 1.0f / (1.0f + e);
#endif
}

// ---- WMMA fragment loads (contiguous k order, same for A and B, wave32) ------------
// A: 16x32 bf16. lanes 0-15: row=lane, elems = K{k0..k0+15}
//               lanes 16-31: row=lane-16, elems = K{k0+16..k0+31}
__device__ __forceinline__ v16bf load_a_frag(const unsigned short* mat /*16 x 256 row-major*/,
                                             int lane, int k0) {
  int half = lane >> 4;
  int row  = lane & 15;
  int kb   = k0 + half * 16;
  union { v16bf v; uint4 q[2]; } a;
  a.q[0] = *(const uint4*)(mat + row * RD + kb);
  a.q[1] = *(const uint4*)(mat + row * RD + kb + 8);
  return a.v;
}

// B: 32x16 bf16 with B[k,n] = W[n0+n, k] (W row-major [256,256]).
// lanes 0-15: col=lane, K = k0..k0+15 ; lanes 16-31: col=lane-16, K = k0+16..k0+31
__device__ __forceinline__ v16bf load_b_frag(const unsigned short* w,
                                             int lane, int n0, int k0) {
  int half = lane >> 4;
  int col  = lane & 15;
  int kk   = k0 + half * 16;
  const uint4* p = (const uint4*)(w + (size_t)(n0 + col) * RD + kk);
  union { v16bf v; uint4 q[2]; } b;
  b.q[0] = p[0];
  b.q[1] = p[1];
  return b.v;
}

__device__ __forceinline__ v8f wmma_bf16(v16bf a, v16bf b, v8f c) {
  return __builtin_amdgcn_wmma_f32_16x16x32_bf16(false, a, false, b,
                                                 (short)0, c, false, false);
}

// Full K=256 reduction as TWO independent 4-deep WMMA chains (halves the
// dependent-WMMA latency chain; combined with one packed f32 add at the end).
__device__ __forceinline__ v8f gemm_k256(const unsigned short* amat,
                                         const unsigned short* bmat,
                                         int lane, int n0) {
  v8f acc0 = {}, acc1 = {};
  #pragma unroll
  for (int k0 = 0; k0 < RD; k0 += 64) {
    acc0 = wmma_bf16(load_a_frag(amat, lane, k0),
                     load_b_frag(bmat, lane, n0, k0), acc0);
    acc1 = wmma_bf16(load_a_frag(amat, lane, k0 + 32),
                     load_b_frag(bmat, lane, n0, k0 + 32), acc1);
  }
  return acc0 + acc1;
}

// ---------------------------------------------------------------------------
// Kernel 0: convert E, Wx, Wz, Wh (f32) -> bf16, packed consecutively in ws
// ---------------------------------------------------------------------------
__global__ void __launch_bounds__(256) rnn_cvt_kernel(
    const float* __restrict__ E, const float* __restrict__ Wx,
    const float* __restrict__ Wz, const float* __restrict__ Wh,
    unsigned short* __restrict__ dst) {
  int i = blockIdx.x * 256 + threadIdx.x;          // 0 .. 262143
  int which = i >> 16, off = i & 65535;
  const float* src = (which == 0) ? E : (which == 1) ? Wx : (which == 2) ? Wz : Wh;
  dst[i] = f2bf(src[off]);
}

// ---------------------------------------------------------------------------
// Kernel 1: embedding gather + wx = x@Wx^T, gate = sigmoid(x@Wz^T)
// Output time-major [T, B, D] (bf16) so the recurrence reads contiguous slabs.
// Grid: (T*B/16) blocks of 256 threads (8 waves). Block = 16 rows (one t, 16 b).
// ---------------------------------------------------------------------------
__global__ void __launch_bounds__(256) rnn_proj_kernel(
    const int* __restrict__ tokens,
    const unsigned short* __restrict__ E_bf,
    const unsigned short* __restrict__ Wx_bf,
    const unsigned short* __restrict__ Wz_bf,
    unsigned short* __restrict__ wx_pre,
    unsigned short* __restrict__ gate_pre) {
  __shared__ __align__(16) unsigned short xs[16 * RD];
  __shared__ int toks[16];

  const int tid  = threadIdx.x;
  const int lane = tid & 31;
  const int wv   = tid >> 5;
  const int R0    = blockIdx.x * 16;   // time-major row index = t*B + b
  const int t     = R0 >> 6;           // R0 / 64
  const int bbase = R0 & 63;

  if (tid < 16) toks[tid] = tokens[(size_t)(bbase + tid) * RT + t];
  __syncthreads();

  // stage x tile [16,256] bf16 into LDS (512 x uint4)
  for (int cid = tid; cid < 512; cid += 256) {
    int row = cid >> 5, off = cid & 31;
    ((uint4*)xs)[cid] = ((const uint4*)(E_bf + (size_t)toks[row] * RD))[off];
  }
  __syncthreads();

  // 32 jobs: 2 matrices x 16 column tiles ; 8 waves x 4 jobs
  for (int jj = 0; jj < 4; ++jj) {
    const int job = wv + jj * 8;
    const int m   = job >> 4;                 // 0 = Wx path, 1 = Wz/gate path
    const int n0  = (job & 15) * 16;
    const unsigned short* W = m ? Wz_bf : Wx_bf;
    v8f acc = gemm_k256(xs, W, lane, n0);
    const int half = lane >> 4;
    const int c    = n0 + (lane & 15);
    unsigned short* dstbuf = m ? gate_pre : wx_pre;
    #pragma unroll
    for (int r = 0; r < 8; ++r) {
      int mm = r + half * 8;
      float v = acc[r];
      if (m) v = fast_sigmoid(v);
      dstbuf[(size_t)(R0 + mm) * RD + c] = f2bf(v);
    }
  }
}

// ---------------------------------------------------------------------------
// Kernel 2: sequential recurrence. 4 blocks x 512 threads (16 waves).
// Block owns batch rows [b0, b0+16). LDS: Wh (128KB) + h ping-pong + wx/gate
// staging = 160KB dynamic. Per step, wave w computes columns [16w, 16w+16).
// wx/gate slabs are staged memory->LDS with ASYNCcnt-tracked DMA loads.
// ---------------------------------------------------------------------------
__global__ void __launch_bounds__(512) rnn_recur_kernel(
    const unsigned short* __restrict__ wh_bf,     // [256,256]
    const unsigned short* __restrict__ wx_pre,    // [T,B,D]
    const unsigned short* __restrict__ gate_pre,  // [T,B,D]
    unsigned short* __restrict__ y_out) {         // [B,T,D]
  extern __shared__ __align__(16) char smem[];
  unsigned short* whs   = (unsigned short*)(smem);            // 131072 B
  unsigned short* hbuf0 = (unsigned short*)(smem + 131072);   //   8192 B
  unsigned short* hbuf1 = (unsigned short*)(smem + 139264);   //   8192 B
  unsigned short* wxs   = (unsigned short*)(smem + 147456);   //   8192 B
  unsigned short* gts   = (unsigned short*)(smem + 155648);   //   8192 B

  const int tid  = threadIdx.x;
  const int lane = tid & 31;
  const int wv   = tid >> 5;            // 0..15
  const int b0   = blockIdx.x * 16;
  const int n0   = wv * 16;

  // preload Wh into LDS: 131072 bytes = 8192 uint4
  {
    const uint4* src = (const uint4*)wh_bf;
    uint4* dst = (uint4*)whs;
    for (int i = tid; i < 8192; i += 512) dst[i] = src[i];
  }
  // h0 = 0 : 8192 bytes = 512 uint4
  {
    uint4 z; z.x = z.y = z.z = z.w = 0u;
    ((uint4*)hbuf0)[tid] = z;
  }
  __syncthreads();

#if RNN_USE_ASYNC_STAGING
  // LDS byte offsets of the staging buffers (generic addr low 32 bits = LDS offset)
  const unsigned int lds_wx = (unsigned int)(size_t)wxs + (unsigned int)tid * 16u;
  const unsigned int lds_gt = (unsigned int)(size_t)gts + (unsigned int)tid * 16u;
  const unsigned long long wx_base = (unsigned long long)(size_t)wx_pre;
  const unsigned long long gt_base = (unsigned long long)(size_t)gate_pre;
#endif

  unsigned short* hcur = hbuf0;
  unsigned short* hnxt = hbuf1;

  for (int t = 0; t < RT; ++t) {
    // stage wx/gate slab for (t, b0..b0+15, :) — contiguous 8KB each
    const size_t so = ((size_t)t * RB + b0) * RD;
#if RNN_USE_ASYNC_STAGING
    {
      // GVS-mode async DMA: LDS[vdst] = MEM[saddr + vaddr]; ASYNCcnt-tracked.
      const unsigned int voff = (unsigned int)(so * 2) + (unsigned int)tid * 16u;
      asm volatile("global_load_async_to_lds_b128 %0, %2, %3\n\t"
                   "global_load_async_to_lds_b128 %1, %2, %4"
                   :
                   : "v"(lds_wx), "v"(lds_gt), "v"(voff), "s"(wx_base), "s"(gt_base)
                   : "memory");
    }
#else
    ((uint4*)wxs)[tid] = ((const uint4*)(wx_pre + so))[tid];
    ((uint4*)gts)[tid] = ((const uint4*)(gate_pre + so))[tid];
#endif

    // prefetch next step's slabs into cache (global_prefetch_b8)
    if (t + 1 < RT) {
      const size_t sn = ((size_t)(t + 1) * RB + b0) * RD + ((size_t)(tid & 255) << 4);
      __builtin_prefetch(wx_pre + sn, 0, 0);
      __builtin_prefetch(gate_pre + sn, 0, 0);
    }

    // H(16x256) @ Wh^T column tile n0 : two 4-deep WMMA chains
    v8f acc = gemm_k256(hcur, whs, lane, n0);

#if RNN_USE_ASYNC_STAGING
    asm volatile("s_wait_asynccnt 0x0" ::: "memory");  // staging DMA landed
#endif
    __syncthreads();   // staging visible to all waves; hcur reads complete

    const int half = lane >> 4;
    const int c    = n0 + (lane & 15);
    #pragma unroll
    for (int r = 0; r < 8; ++r) {
      int m    = r + half * 8;
      float wx = bf2f(wxs[m * RD + c]);
      float g  = bf2f(gts[m * RD + c]);
      float hn = fast_tanh(acc[r] + wx);     // v_tanh_f32
      hnxt[m * RD + c] = f2bf(hn);
      y_out[((size_t)(b0 + m) * RT + t) * RD + c] = f2bf(hn * g);
    }
    __syncthreads();   // hnxt visible for next step
    unsigned short* tmp = hcur; hcur = hnxt; hnxt = tmp;
  }
}

// ---------------------------------------------------------------------------
// Kernel 3: tied head, logits = y @ E^T  ([B*T,256] x [256,256]^T -> f32)
// ---------------------------------------------------------------------------
__global__ void __launch_bounds__(256) rnn_head_kernel(
    const unsigned short* __restrict__ y_bf,   // [B*T, 256]
    const unsigned short* __restrict__ E_bf,   // [256, 256]
    float* __restrict__ out) {                 // [B*T, 256]
  __shared__ __align__(16) unsigned short ys[16 * RD];
  const int tid  = threadIdx.x;
  const int lane = tid & 31;
  const int wv   = tid >> 5;
  const size_t R0 = (size_t)blockIdx.x * 16;

  for (int cid = tid; cid < 512; cid += 256)
    ((uint4*)ys)[cid] = ((const uint4*)(y_bf + R0 * RD))[cid];
  __syncthreads();

  for (int jj = 0; jj < 2; ++jj) {
    const int n0 = (wv + jj * 8) * 16;
    v8f acc = gemm_k256(ys, E_bf, lane, n0);
    const int half = lane >> 4;
    const int c    = n0 + (lane & 15);
    #pragma unroll
    for (int r = 0; r < 8; ++r) {
      int m = r + half * 8;
      // logits written once, never re-read: bypass cache (nontemporal)
      __builtin_nontemporal_store(acc[r], &out[(R0 + m) * RV + c]);
    }
  }
}

// ---------------------------------------------------------------------------
// Host launcher
// ---------------------------------------------------------------------------
extern "C" void kernel_launch(void* const* d_in, const int* in_sizes, int n_in,
                              void* d_out, int out_size, void* d_ws, size_t ws_size,
                              hipStream_t stream) {
  const int*   tokens = (const int*)d_in[0];
  const float* E      = (const float*)d_in[1];
  const float* Wx     = (const float*)d_in[2];
  const float* Wh     = (const float*)d_in[3];
  const float* Wz     = (const float*)d_in[4];
  float* out = (float*)d_out;

  // workspace layout (bf16 elements)
  unsigned short* ws = (unsigned short*)d_ws;
  unsigned short* E_bf     = ws;                        //  65536
  unsigned short* Wx_bf    = ws + 65536;                //  65536
  unsigned short* Wz_bf    = ws + 131072;               //  65536
  unsigned short* Wh_bf    = ws + 196608;               //  65536
  unsigned short* wx_pre   = ws + 262144;               // T*B*D = 33554432
  unsigned short* gate_pre = wx_pre + (size_t)RT * RB * RD;
  unsigned short* y_bf     = gate_pre + (size_t)RT * RB * RD;
  // total: ~192.5 MB of d_ws

  // 0) weight conversion to bf16 (dst packed E,Wx,Wz,Wh)
  rnn_cvt_kernel<<<1024, 256, 0, stream>>>(E, Wx, Wz, Wh, E_bf);

  // 1) projections + gate over all (b,t): 8192 blocks
  rnn_proj_kernel<<<(RT * RB) / 16, 256, 0, stream>>>(
      tokens, E_bf, Wx_bf, Wz_bf, wx_pre, gate_pre);

  // 2) sequential recurrence: 4 blocks (16 batch rows each), 160KB LDS
  rnn_recur_kernel<<<RB / 16, 512, 163840, stream>>>(
      Wh_bf, wx_pre, gate_pre, y_bf);

  // 3) tied-embedding head -> f32 logits
  rnn_head_kernel<<<(RB * RT) / 16, 256, 0, stream>>>(y_bf, E_bf, out);
}
